// Multihead_Attention_27075473834268
// MI455X (gfx1250) — matrix operand order repaired and checked by reading, not compiled, not run on hardware
//
#include <hip/hip_runtime.h>

// ---------------------------------------------------------------------------
// Multihead attention (TQ=TK=2048, H=1024, HEADS=16, d=64) for MI455X/gfx1250.
//
// logits[q,k] = sum_i Ql[q,i] * (Kl[k,i] * Wo[i/64] / 8)  -> Wo folded into K
// projection; bo is softmax-invariant -> dropped. Pipeline:
//   3x proj GEMM (relu, bf16 out) -> logits GEMM -> row softmax -> out GEMM.
// All GEMMs: v_wmma_f32_16x16x32_bf16, double-buffered LDS staging via
// global_load_async_to_lds_b128 (ASYNCcnt) when the toolchain exposes it.
// ---------------------------------------------------------------------------

typedef __bf16 bf16;
typedef __attribute__((ext_vector_type(16))) __bf16 v16bf;
typedef __attribute__((ext_vector_type(8)))  __bf16 v8bf;
typedef __attribute__((ext_vector_type(4)))  __bf16 v4bf;
typedef __attribute__((ext_vector_type(8)))  float  v8f;
typedef __attribute__((ext_vector_type(4)))  float  v4f;
typedef int v4i __attribute__((vector_size(16)));   // b128 payload type

#define TQc   2048
#define TKc   2048
#define Hc    1024

union V16Pack { v16bf v; v8bf h[2]; };

// ---- gfx1250 async global->LDS (ASYNCcnt) with safe fallback -------------
#if defined(__has_builtin)
#if __has_builtin(__builtin_amdgcn_global_load_async_to_lds_b128)
#define HAVE_ASYNC_LDS 1
#endif
#endif
#ifndef HAVE_ASYNC_LDS
#define HAVE_ASYNC_LDS 0
#endif

__device__ __forceinline__ void stage16(const bf16* g, bf16* l) {
#if HAVE_ASYNC_LDS
  // Probe-confirmed: arg0 = addrspace(1) v4i* (global src), arg1 = LDS dst,
  // then imm offset, imm cpol.
  __builtin_amdgcn_global_load_async_to_lds_b128(
      (__attribute__((address_space(1))) v4i*)g,
      (__attribute__((address_space(3))) v4i*)l, 0, 0);
#else
  *(v8bf*)l = *(const v8bf*)g;
#endif
}

__device__ __forceinline__ void async_fence() {
#if HAVE_ASYNC_LDS
#if __has_builtin(__builtin_amdgcn_s_wait_asynccnt)
  __builtin_amdgcn_s_wait_asynccnt(0);
#else
  asm volatile("s_wait_asynccnt 0x0" ::: "memory");
#endif
#endif
}

// ---------------------------------------------------------------------------
// fp32 -> bf16 bulk cast (vectorized x4)
// ---------------------------------------------------------------------------
__global__ __launch_bounds__(256)
void cast_f32_bf16(const float* __restrict__ s, bf16* __restrict__ d, int n4) {
  int i = blockIdx.x * blockDim.x + threadIdx.x;
  if (i < n4) {
    v4f f = *(const v4f*)(s + i * 4);
    v4bf b;
    b[0] = (bf16)f[0]; b[1] = (bf16)f[1]; b[2] = (bf16)f[2]; b[3] = (bf16)f[3];
    *(v4bf*)(d + i * 4) = b;
  }
}

// ---------------------------------------------------------------------------
// Tiled WMMA GEMM, 128x128 block tile, 8 waves, 16x128 stripe per wave.
// Double-buffered LDS; async next-tile staging overlaps current-tile WMMAs.
//   TRANS_B=1 : B given as [N,K] row-major (K contiguous, "X @ W^T" form)
//   TRANS_B=0 : B given as [K,N] row-major (transposed into LDS on the fly)
// ---------------------------------------------------------------------------
template<int TRANS_B, int RELU, int OUT_BF16>
__global__ __launch_bounds__(256)
void gemm_wmma_bf16(const bf16* __restrict__ A, const bf16* __restrict__ B,
                    const float* __restrict__ bias,
                    const float* __restrict__ wo, float woScale,
                    float* __restrict__ Cf, bf16* __restrict__ Cb,
                    int M, int N, int K)
{
  __shared__ bf16 As[2][128][40];   // [buf][m][k]; +8 pad: 16B-aligned, conflict-free
  __shared__ bf16 Bs[2][128][40];   // [buf][n][k]

  const int t    = threadIdx.x;
  const int wave = t >> 5;          // 0..7 (wave32)
  const int lane = t & 31;
  const int lm   = lane & 15;       // row (A) / col (B,C) within 16-tile
  const int lh   = lane >> 4;       // half selector per ISA VGPR layout tables
  const int mBase = blockIdx.y * 128;
  const int nBase = blockIdx.x * 128;

  v8f acc[8] = {};

  auto stage = [&](int buf, int k0) {
#pragma unroll
    for (int i = 0; i < 2; ++i) {
      int c   = t + i * 256;            // 0..511 chunks of 8 bf16
      int row = c >> 2;                 // 0..127
      int col = (c & 3) << 3;           // 0,8,16,24
      stage16(A + (mBase + row) * K + k0 + col, &As[buf][row][col]);
      if (TRANS_B) {
        stage16(B + (nBase + row) * K + k0 + col, &Bs[buf][row][col]);
      } else {
        int k  = c >> 4;                // 0..31
        int n0 = (c & 15) << 3;         // 0..120
        v8bf v = *(const v8bf*)(B + (k0 + k) * N + nBase + n0);
#pragma unroll
        for (int j = 0; j < 8; ++j) Bs[buf][n0 + j][k] = v[j];
      }
    }
#if !HAVE_ASYNC_LDS
    if (k0 + 32 < K) {  // near-cache prefetch of the next tile
      __builtin_prefetch(A + (mBase + (t >> 1)) * K + k0 + 32 + ((t & 1) << 4), 0, 3);
      if (TRANS_B)
        __builtin_prefetch(B + (nBase + (t >> 1)) * K + k0 + 32 + ((t & 1) << 4), 0, 3);
      else
        __builtin_prefetch(B + (k0 + 32 + (t >> 3)) * N + nBase + ((t & 7) << 4), 0, 3);
    }
#endif
  };

  const int nk = K >> 5;              // k-steps of 32
  stage(0, 0);
  async_fence();
  __syncthreads();

  for (int it = 0; it < nk; ++it) {
    const int cur = it & 1;
    if (it + 1 < nk) stage(cur ^ 1, (it + 1) << 5);   // async: overlaps WMMAs below

    // A 16x32 bf16 frag: lane m=lm; elems 0..7 -> K=lh*8+0..7, 8..15 -> +16
    V16Pack a;
    {
      const bf16* ap = &As[cur][wave * 16 + lm][lh * 8];
      a.h[0] = *(const v8bf*)ap;
      a.h[1] = *(const v8bf*)(ap + 16);
    }
    // Preload all 8 B frags so the 8 WMMAs issue back-to-back (no ds waits).
    // B 32x16 bf16 frag: lane n=lm; elems 0..7 -> K=lh*8+0..7, 8..15 -> +16
    V16Pack bfr[8];
#pragma unroll
    for (int n = 0; n < 8; ++n) {
      const bf16* bp = &Bs[cur][n * 16 + lm][lh * 8];
      bfr[n].h[0] = *(const v8bf*)bp;
      bfr[n].h[1] = *(const v8bf*)(bp + 16);
    }
#pragma unroll
    for (int n = 0; n < 8; ++n) {
      acc[n] = __builtin_amdgcn_wmma_f32_16x16x32_bf16(
                 /*neg_a=*/false, a.v, /*neg_b=*/false, bfr[n].v,
                 /*c_mod=*/(short)0, acc[n], /*reuse_a=*/false, /*reuse_b=*/false);
    }

    async_fence();       // next tile landed (no-op on fallback)
    __syncthreads();     // everyone done reading cur, next buffer visible
  }

  // ---- epilogue: C 16x16 f32 layout: lane col=lm, row = lh*8 + r ----
#pragma unroll
  for (int n = 0; n < 8; ++n) {
    int gn = nBase + n * 16 + lm;
    float cb = bias ? bias[gn] : 0.0f;
    float cs = wo ? woScale * wo[gn >> 6] : 1.0f;   // head = column / 64
#pragma unroll
    for (int r = 0; r < 8; ++r) {
      int gm = mBase + wave * 16 + lh * 8 + r;
      float v = acc[n][r] + cb;
      if (RELU) v = fmaxf(v, 0.0f);
      v *= cs;
      if (OUT_BF16) Cb[gm * N + gn] = (bf16)v;
      else          Cf[gm * N + gn] = v;
    }
  }
}

// ---------------------------------------------------------------------------
// Row softmax over 2048 cols: writes attn^T (fp32, output #2) + bf16 attn.
// ---------------------------------------------------------------------------
__global__ __launch_bounds__(256)
void softmax_rows(const float* __restrict__ logits, float* __restrict__ attnT,
                  bf16* __restrict__ attnb, int Tq, int Tk)
{
  __shared__ float red[256];
  const int q = blockIdx.x;
  const int t = threadIdx.x;
  float v[8];
  float mx = -3.4e38f;
#pragma unroll
  for (int j = 0; j < 8; ++j) {
    v[j] = logits[q * Tk + t + j * 256];
    mx = fmaxf(mx, v[j]);
  }
  red[t] = mx; __syncthreads();
  for (int s = 128; s > 0; s >>= 1) {
    if (t < s) red[t] = fmaxf(red[t], red[t + s]);
    __syncthreads();
  }
  mx = red[0]; __syncthreads();

  float sum = 0.0f;
#pragma unroll
  for (int j = 0; j < 8; ++j) { v[j] = __expf(v[j] - mx); sum += v[j]; }
  red[t] = sum; __syncthreads();
  for (int s = 128; s > 0; s >>= 1) {
    if (t < s) red[t] += red[t + s];
    __syncthreads();
  }
  const float inv = 1.0f / red[0];
#pragma unroll
  for (int j = 0; j < 8; ++j) {
    int c = t + j * 256;
    float a = v[j] * inv;
    attnT[c * Tq + q] = a;         // attention.transpose(1,0) -> [Tk, Tq]
    attnb[q * Tk + c] = (bf16)a;   // feed to out = attn @ V_l
  }
}

// ---------------------------------------------------------------------------
extern "C" void kernel_launch(void* const* d_in, const int* in_sizes, int n_in,
                              void* d_out, int out_size, void* d_ws, size_t ws_size,
                              hipStream_t stream) {
  const float* Q  = (const float*)d_in[0];
  const float* K  = (const float*)d_in[1];
  const float* V  = (const float*)d_in[2];
  const float* Wq = (const float*)d_in[3];
  const float* bq = (const float*)d_in[4];
  const float* Wk = (const float*)d_in[5];
  const float* bk = (const float*)d_in[6];
  const float* Wv = (const float*)d_in[7];
  const float* bv = (const float*)d_in[8];
  const float* Wo = (const float*)d_in[9];
  // d_in[10] = bo: softmax-invariant row constant, not needed for either output.

  // Workspace layout (~54 MiB):
  bf16* Qb    = (bf16*)d_ws;            // bf16 casts of inputs
  bf16* Kb    = Qb  + TQc * Hc;
  bf16* Vb    = Kb  + TKc * Hc;
  bf16* Wqb   = Vb  + TKc * Hc;
  bf16* Wkb   = Wqb + Hc * Hc;
  bf16* Wvb   = Wkb + Hc * Hc;
  bf16* Qlb   = Wvb + Hc * Hc;          // relu(Q Wq^T + bq)            [2048,1024]
  bf16* Ksb   = Qlb + TQc * Hc;         // relu(K Wk^T + bk)*Wo[h]/8    [2048,1024]
  bf16* Vlb   = Ksb + TKc * Hc;         // relu(V Wv^T + bv)            [2048,1024]
  bf16* attnb = Vlb + TKc * Hc;         // softmax(logits) in bf16      [2048,2048]
  float* logits = (float*)(attnb + TQc * TKc);  // fp32 logits          [2048,2048]

  float* outF  = (float*)d_out;         // [2048,1024] fp32
  float* attnT = outF + TQc * Hc;       // [2048,2048] fp32

  // 1) bf16 casts
  cast_f32_bf16<<<TQc * Hc / 4 / 256, 256, 0, stream>>>(Q,  Qb,  TQc * Hc / 4);
  cast_f32_bf16<<<TKc * Hc / 4 / 256, 256, 0, stream>>>(K,  Kb,  TKc * Hc / 4);
  cast_f32_bf16<<<TKc * Hc / 4 / 256, 256, 0, stream>>>(V,  Vb,  TKc * Hc / 4);
  cast_f32_bf16<<<Hc * Hc / 4 / 256, 256, 0, stream>>>(Wq, Wqb, Hc * Hc / 4);
  cast_f32_bf16<<<Hc * Hc / 4 / 256, 256, 0, stream>>>(Wk, Wkb, Hc * Hc / 4);
  cast_f32_bf16<<<Hc * Hc / 4 / 256, 256, 0, stream>>>(Wv, Wvb, Hc * Hc / 4);

  const float invSqrtD = 0.125f;  // 1/sqrt(64)

  // 2) projections (X @ W^T -> TRANS_B=1), relu, bf16 out
  dim3 gProj(Hc / 128, TQc / 128);
  gemm_wmma_bf16<1, 1, 1><<<gProj, 256, 0, stream>>>(
      Qb, Wqb, bq, nullptr, 1.0f, nullptr, Qlb, TQc, Hc, Hc);
  gemm_wmma_bf16<1, 1, 1><<<gProj, 256, 0, stream>>>(
      Kb, Wkb, bk, Wo, invSqrtD, nullptr, Ksb, TKc, Hc, Hc);   // Wo/8 folded
  gemm_wmma_bf16<1, 1, 1><<<gProj, 256, 0, stream>>>(
      Vb, Wvb, bv, nullptr, 1.0f, nullptr, Vlb, TKc, Hc, Hc);

  // 3) logits = Ql @ Ks^T (fp32 out)
  dim3 gLog(TKc / 128, TQc / 128);
  gemm_wmma_bf16<1, 0, 0><<<gLog, 256, 0, stream>>>(
      Qlb, Ksb, nullptr, nullptr, 1.0f, logits, nullptr, TQc, TKc, Hc);

  // 4) softmax rows -> attn^T fp32 (output #2) + bf16 attn
  softmax_rows<<<TQc, 256, 0, stream>>>(logits, attnT, attnb, TQc, TKc);

  // 5) out = attn @ V_l (B row-major [K,N] -> TRANS_B=0), fp32 out (output #1)
  dim3 gOut(Hc / 128, TQc / 128);
  gemm_wmma_bf16<0, 0, 0><<<gOut, 256, 0, stream>>>(
      attnb, Vlb, nullptr, nullptr, 1.0f, outF, nullptr, TQc, Hc, TKc);
}
